// LSTMModelBase_19138374270976
// MI455X (gfx1250) — compile-verified, matrix operand order checked by reading, not run on hardware
//
#include <hip/hip_runtime.h>
#include <hip/hip_bf16.h>
#include <stdint.h>

// ---------------------------------------------------------------------------
// LSTM scan on MI455X (gfx1250): bf16 WMMA GEMMs (v_wmma_f32_16x16x32_bf16),
// fp32 recurrent state, fused gate/cell update, per-step kernel pair.
// Round 2: software-pipelined A/B fragment loads (hide ds/global latency
// behind WMMA issue), B-operand reuse hints, near-cache weight prefetch.
// ---------------------------------------------------------------------------

typedef __attribute__((ext_vector_type(16))) __bf16        v16bf;
typedef __attribute__((ext_vector_type(8)))  float         v8f;
typedef __attribute__((ext_vector_type(4)))  unsigned int  u32x4;

union FragBF { v16bf v; u32x4 q[2]; };

static constexpr int BB = 64;    // batch == GEMM M
static constexpr int SS = 512;   // sequence length
static constexpr int DD = 1024;  // input dim == GEMM K
static constexpr int HH = 1024;  // hidden dim == GEMM N
static constexpr int KT = DD / 32;      // 32 k-tiles of depth 32
static constexpr int ZPITCH = DD + 8;   // bf16 elems; 16B skew to spread LDS banks

__device__ __forceinline__ float sigm_f(float x) {
  return 1.0f / (1.0f + __expf(-x));
}
__device__ __forceinline__ float tanh_f(float x) {
  return 1.0f - 2.0f / (__expf(2.0f * x) + 1.0f);
}

// Stage a contiguous 64x1024 bf16 matrix from global into LDS with skewed pitch.
__device__ __forceinline__ void stage_64x1024(const __bf16* __restrict__ g,
                                              __bf16* __restrict__ lds) {
  const u32x4* gs = (const u32x4*)g;                   // 128 u32x4 per row
  for (int c = threadIdx.x; c < BB * 128; c += blockDim.x) {
    int row = c >> 7;
    int col = c & 127;
    *(u32x4*)(lds + row * ZPITCH + col * 8) = gs[c];
  }
}

// ---------------------------------------------------------------------------
// Weight repack: f32 (K x N, row-major) -> bf16 WMMA B-fragments.
// Per-lane element j of a 16x16x32 bf16 B-fragment maps to
//   n = n_tile*16 + (lane & 15)
//   k = k_tile*32 + ((j>>3)<<4) + ((lane>>4)<<3) + (j & 7)
// Packed layout: [h_tile][gate][k_tile][lane][16] (gate dim only for fused gates)
// ---------------------------------------------------------------------------
__global__ void pack_gate_weights(const float* __restrict__ Wi1,
                                  const float* __restrict__ Wi2,
                                  const float* __restrict__ Wf,
                                  const float* __restrict__ Wo1,
                                  __bf16* __restrict__ dst) {
  int p = blockIdx.x * blockDim.x + threadIdx.x;     // 4*1024*1024 elems
  if (p >= 4 * DD * HH) return;
  int j    =  p        & 15;
  int lane = (p >> 4)  & 31;
  int kt   = (p >> 9)  & 31;
  int gate = (p >> 14) & 3;
  int ht   =  p >> 16;
  int n = ht * 16 + (lane & 15);
  int k = kt * 32 + ((j >> 3) << 4) + ((lane >> 4) << 3) + (j & 7);
  const float* W = (gate == 0) ? Wi1 : (gate == 1) ? Wi2 : (gate == 2) ? Wf : Wo1;
  dst[p] = (__bf16)W[(size_t)k * HH + n];
}

__global__ void pack_out_weights(const float* __restrict__ W2,
                                 __bf16* __restrict__ dst) {
  int p = blockIdx.x * blockDim.x + threadIdx.x;     // 1024*1024 elems
  if (p >= DD * HH) return;
  int j    =  p       & 15;
  int lane = (p >> 4) & 31;
  int kt   = (p >> 9) & 31;
  int ht   =  p >> 14;
  int n = ht * 16 + (lane & 15);
  int k = kt * 32 + ((j >> 3) << 4) + ((lane >> 4) << 3) + (j & 7);
  dst[p] = (__bf16)W2[(size_t)k * HH + n];
}

// Z0 = bf16(X[:,0,:]);  Ct = 0
__global__ void init_state(const float* __restrict__ X,
                           __bf16* __restrict__ Zbuf,
                           float* __restrict__ Ct) {
  int idx = blockIdx.x * blockDim.x + threadIdx.x;   // 64*1024
  if (idx >= BB * DD) return;
  int b = idx >> 10, d = idx & 1023;
  Zbuf[idx] = (__bf16)X[(size_t)b * SS * DD + d];
  Ct[idx]   = 0.0f;
}

// Software-pipelined 64x16 x (K=1024) WMMA GEMM over packed-B weights and an
// LDS-staged 64x1024 bf16 A matrix. acc[mt] covers rows mt*16..mt*16+15.
__device__ __forceinline__ void gemm_64x16(const __bf16* __restrict__ ash,
                                           const u32x4* __restrict__ wbase,
                                           int lane, v8f acc[4]) {
  const int ln = lane & 15;
  const int lh = lane >> 4;

  FragBF bcur, bnxt;
  FragBF acur[4], anxt[4];

  // prologue: k-tile 0
  bcur.q[0] = wbase[lane * 2 + 0];
  bcur.q[1] = wbase[lane * 2 + 1];
  {
    const int k0 = lh * 8;
#pragma unroll
    for (int mt = 0; mt < 4; ++mt) {
      const int m = mt * 16 + ln;
      acur[mt].q[0] = *(const u32x4*)(ash + m * ZPITCH + k0);
      acur[mt].q[1] = *(const u32x4*)(ash + m * ZPITCH + k0 + 16);
    }
  }

#pragma unroll 2
  for (int kt = 0; kt < KT; ++kt) {
    const int ktn = (kt + 1) & (KT - 1);       // branchless wrap on last iter
    __builtin_prefetch(wbase + ((kt + 2) & (KT - 1)) * 64 + lane * 2, 0, 3);

    // issue next tile's loads first: waits overlap current tile's WMMAs
    bnxt.q[0] = wbase[ktn * 64 + lane * 2 + 0];
    bnxt.q[1] = wbase[ktn * 64 + lane * 2 + 1];
    const int k0 = ktn * 32 + lh * 8;
#pragma unroll
    for (int mt = 0; mt < 4; ++mt) {
      const int m = mt * 16 + ln;
      anxt[mt].q[0] = *(const u32x4*)(ash + m * ZPITCH + k0);
      anxt[mt].q[1] = *(const u32x4*)(ash + m * ZPITCH + k0 + 16);
    }

    // 4 WMMAs sharing B: set RB reuse hint on all but the last
    acc[0] = __builtin_amdgcn_wmma_f32_16x16x32_bf16(
        false, acur[0].v, false, bcur.v, (short)0, acc[0], false, true);
    acc[1] = __builtin_amdgcn_wmma_f32_16x16x32_bf16(
        false, acur[1].v, false, bcur.v, (short)0, acc[1], false, true);
    acc[2] = __builtin_amdgcn_wmma_f32_16x16x32_bf16(
        false, acur[2].v, false, bcur.v, (short)0, acc[2], false, true);
    acc[3] = __builtin_amdgcn_wmma_f32_16x16x32_bf16(
        false, acur[3].v, false, bcur.v, (short)0, acc[3], false, false);

    bcur = bnxt;
#pragma unroll
    for (int mt = 0; mt < 4; ++mt) acur[mt] = anxt[mt];
  }
}

// ---------------------------------------------------------------------------
// Kernel 1 (per step): 4 gate GEMMs (M=64,N=1024,K=1024 fused over gates),
// activations, and the cell update Ct = Ct*f + i1*i2.
// Grid: 32 WGs x 256 threads (8 waves). Wave w -> (h_sub = w>>2, gate = w&3).
// ---------------------------------------------------------------------------
__global__ void lstm_gates(const __bf16* __restrict__ Zbuf,
                           const __bf16* __restrict__ Wp,
                           const float* __restrict__ b_i1,
                           const float* __restrict__ b_i2,
                           const float* __restrict__ b_f,
                           const float* __restrict__ b_o1,
                           float* __restrict__ Ct,
                           __bf16* __restrict__ Ctbf,
                           float* __restrict__ o1ws) {
  extern __shared__ __align__(16) char smem[];
  __bf16* zsh = (__bf16*)smem;                               // 64 x ZPITCH bf16
  float*  gsh = (float*)(smem + (size_t)BB * ZPITCH * 2);    // [2][4][64][16] f32

  stage_64x1024(Zbuf, zsh);
  __syncthreads();

  const int lane = threadIdx.x & 31;
  const int wave = threadIdx.x >> 5;
  const int hsub = wave >> 2;
  const int gate = wave & 3;
  const int ht   = blockIdx.x * 2 + hsub;
  const int ln   = lane & 15;
  const int lh   = lane >> 4;

  v8f acc[4] = {v8f{}, v8f{}, v8f{}, v8f{}};
  const u32x4* wbase = (const u32x4*)Wp + (size_t)(ht * 4 + gate) * KT * 64;

  gemm_64x16(zsh, wbase, lane, acc);

  // activation epilogue -> LDS staging
  const int n = ht * 16 + ln;
  const float bias = (gate == 0) ? b_i1[n] : (gate == 1) ? b_i2[n]
                   : (gate == 2) ? b_f[n]  : b_o1[n];
#pragma unroll
  for (int mt = 0; mt < 4; ++mt) {
#pragma unroll
    for (int r = 0; r < 8; ++r) {
      float x = acc[mt][r] + bias;
      float a = (gate == 1) ? tanh_f(x) : sigm_f(x);
      int m = mt * 16 + r + lh * 8;
      gsh[((hsub * 4 + gate) * BB + m) * 16 + ln] = a;
    }
  }
  __syncthreads();

  // fused cell update for this WG's 32 columns
  for (int idx = threadIdx.x; idx < 2 * BB * 16; idx += blockDim.x) {
    int hs  = idx >> 10;
    int rem = idx & 1023;
    int m   = rem >> 4;
    int c   = rem & 15;
    int nn  = (blockIdx.x * 2 + hs) * 16 + c;
    const float* gb = gsh + (size_t)(hs * 4) * BB * 16;
    float i1 = gb[(0 * BB + m) * 16 + c];
    float i2 = gb[(1 * BB + m) * 16 + c];
    float f  = gb[(2 * BB + m) * 16 + c];
    float o1 = gb[(3 * BB + m) * 16 + c];
    float ct = Ct[m * HH + nn] * f + i1 * i2;
    Ct[m * HH + nn]   = ct;
    Ctbf[m * HH + nn] = (__bf16)ct;
    o1ws[m * HH + nn] = o1;
  }
}

// ---------------------------------------------------------------------------
// Kernel 2 (per step): o2 = tanh(Ct @ W_output2 + b), Ht = o1*o2 -> d_out,
// and next-step Z = bf16(Ht + X[:,t+1,:]).
// Grid: 8 WGs x 256 threads (8 waves), wave -> one h-tile.
// ---------------------------------------------------------------------------
__global__ void lstm_out(const __bf16* __restrict__ Ctbf,
                         const __bf16* __restrict__ Wp2,
                         const float* __restrict__ b_o2,
                         const float* __restrict__ o1ws,
                         const float* __restrict__ X,
                         __bf16* __restrict__ Zbuf,
                         float* __restrict__ hidden,
                         int t) {
  extern __shared__ __align__(16) char smem[];
  __bf16* csh = (__bf16*)smem;                         // 64 x ZPITCH bf16

  stage_64x1024(Ctbf, csh);
  __syncthreads();

  const int lane = threadIdx.x & 31;
  const int wave = threadIdx.x >> 5;
  const int ht   = blockIdx.x * 8 + wave;
  const int ln   = lane & 15;
  const int lh   = lane >> 4;

  v8f acc[4] = {v8f{}, v8f{}, v8f{}, v8f{}};
  const u32x4* wbase = (const u32x4*)Wp2 + (size_t)ht * KT * 64;

  gemm_64x16(csh, wbase, lane, acc);

  const int n = ht * 16 + ln;
  const float bias = b_o2[n];
#pragma unroll
  for (int mt = 0; mt < 4; ++mt) {
#pragma unroll
    for (int r = 0; r < 8; ++r) {
      int m = mt * 16 + r + lh * 8;
      float o2 = tanh_f(acc[mt][r] + bias);
      float h  = o1ws[m * HH + n] * o2;
      hidden[((size_t)m * SS + t) * HH + n] = h;
      if (t + 1 < SS) {
        float xn = X[((size_t)m * SS + (t + 1)) * DD + n];
        Zbuf[m * DD + n] = (__bf16)(h + xn);
      }
    }
  }
}

// Ht = hidden[:, S-1, :]; Ct copied from state -> output tail
__global__ void finalize(const float* __restrict__ hidden,
                         const float* __restrict__ Ct,
                         float* __restrict__ outHt,
                         float* __restrict__ outCt) {
  int idx = blockIdx.x * blockDim.x + threadIdx.x;  // 64*1024
  if (idx >= BB * HH) return;
  int b = idx >> 10, n = idx & 1023;
  outHt[idx] = hidden[((size_t)b * SS + (SS - 1)) * HH + n];
  outCt[idx] = Ct[idx];
}

// ---------------------------------------------------------------------------
extern "C" void kernel_launch(void* const* d_in, const int* in_sizes, int n_in,
                              void* d_out, int out_size, void* d_ws, size_t ws_size,
                              hipStream_t stream) {
  const float* X    = (const float*)d_in[0];
  const float* Wi1  = (const float*)d_in[1];
  const float* bi1  = (const float*)d_in[2];
  const float* Wi2  = (const float*)d_in[3];
  const float* bi2  = (const float*)d_in[4];
  const float* Wf   = (const float*)d_in[5];
  const float* bf_  = (const float*)d_in[6];
  const float* Wo1  = (const float*)d_in[7];
  const float* bo1  = (const float*)d_in[8];
  const float* W2   = (const float*)d_in[9];
  const float* bo2  = (const float*)d_in[10];

  float* hidden = (float*)d_out;                         // (B,S,H)
  float* outHt  = hidden + (size_t)BB * SS * HH;
  float* outCt  = outHt + (size_t)BB * HH;

  // workspace layout
  char* ws = (char*)d_ws;
  __bf16* WpGates = (__bf16*)(ws);                               // 8 MB
  __bf16* Wp2     = (__bf16*)(ws + (size_t)8 * 1024 * 1024);     // 2 MB
  __bf16* Zbuf    = (__bf16*)(ws + (size_t)10 * 1024 * 1024);    // 128 KB
  float*  Ct      = (float*)(ws + (size_t)10 * 1024 * 1024 + 131072);        // 256 KB
  __bf16* Ctbf    = (__bf16*)(ws + (size_t)10 * 1024 * 1024 + 131072 + 262144); // 128 KB
  float*  o1ws    = (float*)(ws + (size_t)10 * 1024 * 1024 + 131072 + 262144 + 131072); // 256 KB

  const size_t smem1 = (size_t)BB * ZPITCH * 2 + (size_t)2 * 4 * BB * 16 * 4;
  const size_t smem2 = (size_t)BB * ZPITCH * 2;

  pack_gate_weights<<<(4 * DD * HH) / 256, 256, 0, stream>>>(Wi1, Wi2, Wf, Wo1, WpGates);
  pack_out_weights<<<(DD * HH) / 256, 256, 0, stream>>>(W2, Wp2);
  init_state<<<(BB * DD) / 256, 256, 0, stream>>>(X, Zbuf, Ct);

  for (int t = 0; t < SS; ++t) {
    lstm_gates<<<32, 256, smem1, stream>>>(Zbuf, WpGates, bi1, bi2, bf_, bo1,
                                           Ct, Ctbf, o1ws);
    lstm_out<<<8, 256, smem2, stream>>>(Ctbf, Wp2, bo2, o1ws, X, Zbuf, hidden, t);
  }

  finalize<<<(BB * HH) / 256, 256, 0, stream>>>(hidden, Ct, outHt, outCt);
}
